// Net_44633300140087
// MI455X (gfx1250) — matrix operand order repaired and checked by reading, not compiled, not run on hardware
//
#include <hip/hip_runtime.h>

typedef __attribute__((ext_vector_type(16))) _Float16 v16h;
typedef __attribute__((ext_vector_type(8)))  float    v8f;

constexpr int   kB     = 256;
constexpr int   kN     = 128;
constexpr float kDRN2  = 0.4f * 0.4f;
constexpr float kSlope = 0.01f;
constexpr float kTwoPi = 6.28318530717958647692f;

struct ConvW { const float *W0, *b0, *W1, *b1, *W2, *b2; };

// ---------------------------------------------------------------------------
// B-fragment for v_wmma_f32_16x16x32_f16:
//   lane L holds column n = L%16; half h -> k = (h<8?h:h+8) + (L>=16 ? 8 : 0).
//   Pad with zeros beyond (K, Ncols).
// ---------------------------------------------------------------------------
__device__ inline v16h make_bfrag(const float* __restrict__ W, int K, int Ncols, int lane)
{
  const int n     = lane & 15;
  const int kbase = (lane & 16) ? 8 : 0;
  v16h f;
#pragma unroll
  for (int h = 0; h < 16; ++h) {
    const int k = kbase + ((h < 8) ? h : h + 8);
    float v = 0.0f;
    if (k < K && n < Ncols) v = W[k * Ncols + n];
    f[h] = (_Float16)v;
  }
  return f;
}

// ---------------------------------------------------------------------------
// One EdgeConv: for every (i,j) pair run MLP (3*KF+2 -> 5 -> 5 -> HOUT) via
// WMMA, then adjacency-masked mean-aggregate into s_agg / s_deg.
// Wave w owns j in [16w, 16w+16); A-fragment row m = pair (i, 16w+m).
// A layout: lane L row m=L%16; half h -> k = (h<8?h:h+8) + (L>=16 ? 8 : 0).
// ---------------------------------------------------------------------------
template <int KF, int HOUT>
__device__ void edge_conv_phase(const ConvW cw,
                                const float* s_pt, const float* s_cx, const float* s_cy,
                                const float (*s_feat)[4], const int* s_valid,
                                float (*s_agg)[5], float* s_deg,
                                float (*st)[8])
{
  constexpr int IN = 3 * KF + 2;
  constexpr int D1 = 5;
  constexpr int D2 = 5;
  const int  tid  = threadIdx.x;
  const int  lane = tid & 31;
  const int  lm   = lane & 15;
  const bool hi   = (lane & 16) != 0;
  const int  wv   = tid >> 5;
  const int  j    = wv * 16 + lm;

  const v16h B0 = make_bfrag(cw.W0, IN, D1, lane);
  const v16h B1 = make_bfrag(cw.W1, D1, D2, lane);
  const v16h B2 = make_bfrag(cw.W2, D2, HOUT, lane);
  const float bb0 = (lm < D1)   ? cw.b0[lm] : 0.0f;
  const float bb1 = (lm < D2)   ? cw.b1[lm] : 0.0f;
  const float bb2 = (lm < HOUT) ? cw.b2[lm] : 0.0f;

  const float cxj = s_cx[j], cyj = s_cy[j];
  const int   vj  = s_valid[j];
  const float r2j = cxj * cxj + cyj * cyj;
  float fj[KF];
#pragma unroll
  for (int k = 0; k < KF; ++k) fj[k] = s_feat[j][k];

  for (int i = 0; i < kN; ++i) {
    const float cxi = s_cx[i], cyi = s_cy[i];
    const int   vi  = s_valid[i];
    const float r2i = cxi * cxi + cyi * cyi;

    float msg[16];
#pragma unroll
    for (int k = 0; k < 16; ++k) msg[k] = 0.0f;
#pragma unroll
    for (int k = 0; k < KF; ++k) {
      const float fi = s_feat[i][k];
      msg[k]          = fi;
      msg[KF + k]     = fj[k];
      msg[2 * KF + k] = fj[k] - fi;
    }
    const float nrm  = sqrtf(r2i * r2j);
    const float safe = (nrm > 0.0f) ? nrm : 1.0f;
    msg[3 * KF]     = (cxi * cxj + cyi * cyj) / safe;
    msg[3 * KF + 1] = (cyi * cxj - cxi * cyj) / safe;

    const float dx = cxi - cxj, dy = cyi - cyj;
    const bool adj = (dx * dx + dy * dy <= kDRN2) && vi && vj;
    const unsigned mask = (unsigned)__ballot(adj) & 0xFFFFu;

    // ---- layer 1: (16 x IN) @ (IN x D1), K padded to 32 ----
    v16h a;
#pragma unroll
    for (int h = 0; h < 16; ++h) {
      const int kc = (h < 8) ? h : (h + 8);
      const float vlo = (kc     < IN) ? msg[kc]     : 0.0f;
      const float vhi = (kc + 8 < IN) ? msg[kc + 8] : 0.0f;
      a[h] = (_Float16)(hi ? vhi : vlo);
    }
    v8f c = {};
    c = __builtin_amdgcn_wmma_f32_16x16x32_f16(false, a, false, B0, (short)0, c, false, false);
#pragma unroll
    for (int r = 0; r < 8; ++r) {
      const float v = c[r] + bb0;
      c[r] = (v >= 0.0f) ? v : kSlope * v;
    }
    if (lm < 8) {
#pragma unroll
      for (int r = 0; r < 8; ++r) st[r + (hi ? 8 : 0)][lm] = c[r];
    }
    __syncthreads();

    // ---- layer 2: (16 x D1) @ (D1 x D2) ----
    v16h a2;
#pragma unroll
    for (int h = 0; h < 16; ++h) {
      const int kc = (h < 8) ? h : (h + 8);
      const float vlo = (kc     < D1) ? st[lm][kc]     : 0.0f;
      const float vhi = (kc + 8 < D1) ? st[lm][kc + 8] : 0.0f;
      a2[h] = (_Float16)(hi ? vhi : vlo);
    }
    __syncthreads();
    v8f c2 = {};
    c2 = __builtin_amdgcn_wmma_f32_16x16x32_f16(false, a2, false, B1, (short)0, c2, false, false);
#pragma unroll
    for (int r = 0; r < 8; ++r) {
      const float v = c2[r] + bb1;
      c2[r] = (v >= 0.0f) ? v : kSlope * v;
    }
    if (lm < 8) {
#pragma unroll
      for (int r = 0; r < 8; ++r) st[r + (hi ? 8 : 0)][lm] = c2[r];
    }
    __syncthreads();

    // ---- layer 3: (16 x D2) @ (D2 x HOUT), no activation ----
    v16h a3;
#pragma unroll
    for (int h = 0; h < 16; ++h) {
      const int kc = (h < 8) ? h : (h + 8);
      const float vlo = (kc     < D2) ? st[lm][kc]     : 0.0f;
      const float vhi = (kc + 8 < D2) ? st[lm][kc + 8] : 0.0f;
      a3[h] = (_Float16)(hi ? vhi : vlo);
    }
    __syncthreads();
    v8f c3 = {};
    c3 = __builtin_amdgcn_wmma_f32_16x16x32_f16(false, a3, false, B2, (short)0, c3, false, false);

    // ---- adjacency-masked sum over the 16 rows (j values) of this tile ----
    float s = 0.0f;
#pragma unroll
    for (int r = 0; r < 8; ++r) {
      const int m = r + (hi ? 8 : 0);
      const float v = c3[r] + bb2;
      s += (mask & (1u << m)) ? v : 0.0f;
    }
    s += __shfl_xor(s, 16, 32);              // fold M-halves (lane L <-> L+16)
    if (!hi && lm < HOUT) atomicAdd(&s_agg[i][lm], s);
    if (lane == 0) atomicAdd(&s_deg[i], (float)__popc(mask));
  }
}

template <int HOUT>
__device__ void node_update(float* s_pt, float* s_cx, float* s_cy,
                            float (*s_feat)[4], float (*s_agg)[5], float* s_deg)
{
  const int t = threadIdx.x;
  if (t < kN) {
    const float deg = s_deg[t];
    const float dv  = fmaxf(deg, 1.0f);
    float ag[HOUT];
#pragma unroll
    for (int n = 0; n < HOUT; ++n) ag[n] = s_agg[t][n] / dv;
    const bool  has  = deg > 0.0f;
    const float pt_o = has ? s_pt[t] : 0.0f;
    const float cox  = has ? s_cx[t] : 0.0f;
    const float coy  = has ? s_cy[t] : 0.0f;
    const float phi  = kTwoPi * ag[HOUT - 1];
    float sp, cp;
    sincosf(phi, &sp, &cp);
    s_pt[t] = pt_o;
#pragma unroll
    for (int n = 0; n < HOUT - 1; ++n) s_feat[t][n] = ag[n];
    s_cx[t] = cp * cox - sp * coy;
    s_cy[t] = sp * cox + cp * coy;
    s_deg[t] = 0.0f;
#pragma unroll
    for (int n = 0; n < 5; ++n) s_agg[t][n] = 0.0f;
  }
}

// ---------------------------------------------------------------------------
// One block per graph: both EdgeConvs + pooling fully resident in LDS.
// ---------------------------------------------------------------------------
__global__ __launch_bounds__(256) void fused_edgeconv_kernel(
    const float* __restrict__ pt, const float* __restrict__ ang,
    const ConvW c0, const ConvW c1,
    float* __restrict__ gws, float* __restrict__ out)
{
  __shared__ float s_pt[kN], s_cx[kN], s_cy[kN], s_deg[kN], s_red[kN];
  __shared__ int   s_valid[kN];
  __shared__ float s_feat[kN][4];
  __shared__ float s_agg[kN][5];
  __shared__ float s_stage[8][16][8];   // per-wave inter-layer transpose tile

  const int b = blockIdx.x;
  const int t = threadIdx.x;

  if (t < kN) {
    const float a0 = ang[(b * kN + t) * 2 + 0];
    const float a1 = ang[(b * kN + t) * 2 + 1];
    s_pt[t] = pt[b * kN + t];
    s_cx[t] = a0;
    s_cy[t] = a1;
    s_feat[t][0] = sqrtf(a0 * a0 + a1 * a1);
    s_feat[t][1] = 0.0f; s_feat[t][2] = 0.0f; s_feat[t][3] = 0.0f;
    s_valid[t] = (fabsf(a0) + fabsf(a1)) != 0.0f;
    s_deg[t] = 0.0f;
#pragma unroll
    for (int n = 0; n < 5; ++n) s_agg[t][n] = 0.0f;
  }
  __syncthreads();

  edge_conv_phase<1, 5>(c0, s_pt, s_cx, s_cy, s_feat, s_valid, s_agg, s_deg, s_stage[t >> 5]);
  __syncthreads();
  node_update<5>(s_pt, s_cx, s_cy, s_feat, s_agg, s_deg);
  __syncthreads();

  edge_conv_phase<4, 3>(c1, s_pt, s_cx, s_cy, s_feat, s_valid, s_agg, s_deg, s_stage[t >> 5]);
  __syncthreads();
  node_update<3>(s_pt, s_cx, s_cy, s_feat, s_agg, s_deg);
  __syncthreads();

  // ---- pt-weighted global pooling ----
  if (t < kN) s_red[t] = s_valid[t] ? s_pt[t] : 0.0f;
  __syncthreads();
  for (int s2 = 64; s2 > 0; s2 >>= 1) {
    if (t < s2) s_red[t] += s_red[t + s2];
    __syncthreads();
  }
  const float tot  = s_red[0];
  const float tinv = 1.0f / ((tot > 0.0f) ? tot : 1.0f);
  __syncthreads();
  if (t < kN) {
    const float w = (s_valid[t] ? s_pt[t] : 0.0f) * tinv;
    s_agg[t][0] = w * s_feat[t][0];
    s_agg[t][1] = w * s_feat[t][1];
    s_agg[t][2] = w * s_cx[t];
    s_agg[t][3] = w * s_cy[t];
  }
  __syncthreads();
  for (int s2 = 64; s2 > 0; s2 >>= 1) {
    if (t < s2) {
#pragma unroll
      for (int f = 0; f < 4; ++f) s_agg[t][f] += s_agg[t + s2][f];
    }
    __syncthreads();
  }
  if (t == 0) {
    gws[b * 4 + 0] = s_agg[0][0];
    gws[b * 4 + 1] = s_agg[0][1];
    out[b * 3 + 1] = s_agg[0][2];   // rotated coords pass straight through
    out[b * 3 + 2] = s_agg[0][3];
  }
}

// ---------------------------------------------------------------------------
// Readout MLP 2 -> 32 -> 32 -> 1 over all 256 graphs: pure WMMA GEMM chain.
// One 16-row M-tile per wave (16 waves, one block).
// ---------------------------------------------------------------------------
__global__ __launch_bounds__(512) void readout_kernel(
    const float* __restrict__ g,
    const float* __restrict__ R0, const float* __restrict__ rb0,
    const float* __restrict__ R1, const float* __restrict__ rb1,
    const float* __restrict__ R2, const float* __restrict__ rb2,
    float* __restrict__ out)
{
  __shared__ float s_act[16][16][32];
  const int  t    = threadIdx.x;
  const int  lane = t & 31;
  const int  wv   = t >> 5;
  const int  lm   = lane & 15;
  const bool hi   = (lane & 16) != 0;
  const int  m0   = wv * 16;
  float (*act)[32] = s_act[wv];

  // ---- layer 1: (16 x 2) @ (2 x 32) ----
  const int   row = m0 + lm;
  const float g0  = g[row * 4 + 0];
  const float g1  = g[row * 4 + 1];
  v16h a1;
#pragma unroll
  for (int h = 0; h < 16; ++h) a1[h] = (_Float16)0.0f;
  a1[0] = (_Float16)(hi ? 0.0f : g0);
  a1[1] = (_Float16)(hi ? 0.0f : g1);

#pragma unroll
  for (int nh = 0; nh < 2; ++nh) {
    const int n = nh * 16 + lm;
    v16h bf;
#pragma unroll
    for (int h = 0; h < 16; ++h) {
      const int k = (hi ? 16 : 0) + h;
      bf[h] = (_Float16)((k < 2) ? R0[k * 32 + n] : 0.0f);
    }
    v8f c = {};
    c = __builtin_amdgcn_wmma_f32_16x16x32_f16(false, a1, false, bf, (short)0, c, false, false);
    const float bias = rb0[n];
#pragma unroll
    for (int r = 0; r < 8; ++r) {
      float v = c[r] + bias;
      v = (v >= 0.0f) ? v : kSlope * v;
      act[r + (hi ? 8 : 0)][n] = v;
    }
  }
  __syncthreads();

  // ---- layer 2: (16 x 32) @ (32 x 32) ----
  v16h a2;
#pragma unroll
  for (int h = 0; h < 16; ++h) {
    const int kc = (h < 8) ? h : (h + 8);
    a2[h] = (_Float16)act[lm][kc + (hi ? 8 : 0)];
  }
  __syncthreads();
#pragma unroll
  for (int nh = 0; nh < 2; ++nh) {
    const int n = nh * 16 + lm;
    v16h bf;
#pragma unroll
    for (int h = 0; h < 16; ++h) {
      const int k = (hi ? 8 : 0) + ((h < 8) ? h : h + 8);
      bf[h] = (_Float16)R1[k * 32 + n];
    }
    v8f c = {};
    c = __builtin_amdgcn_wmma_f32_16x16x32_f16(false, a2, false, bf, (short)0, c, false, false);
    const float bias = rb1[n];
#pragma unroll
    for (int r = 0; r < 8; ++r) {
      float v = c[r] + bias;
      v = (v >= 0.0f) ? v : kSlope * v;
      act[r + (hi ? 8 : 0)][n] = v;
    }
  }
  __syncthreads();

  // ---- layer 3: (16 x 32) @ (32 x 1), no activation ----
  v16h a3;
#pragma unroll
  for (int h = 0; h < 16; ++h) {
    const int kc = (h < 8) ? h : (h + 8);
    a3[h] = (_Float16)act[lm][kc + (hi ? 8 : 0)];
  }
  v16h bf3;
#pragma unroll
  for (int h = 0; h < 16; ++h) {
    const int k = (hi ? 8 : 0) + ((h < 8) ? h : h + 8);
    bf3[h] = (_Float16)((lm == 0) ? R2[k] : 0.0f);
  }
  v8f c = {};
  c = __builtin_amdgcn_wmma_f32_16x16x32_f16(false, a3, false, bf3, (short)0, c, false, false);
  if (lm == 0) {
    const float bias = rb2[0];
#pragma unroll
    for (int r = 0; r < 8; ++r) {
      const int m = r + (hi ? 8 : 0);
      out[(m0 + m) * 3 + 0] = c[r] + bias;
    }
  }
}

extern "C" void kernel_launch(void* const* d_in, const int* in_sizes, int n_in,
                              void* d_out, int out_size, void* d_ws, size_t ws_size,
                              hipStream_t stream)
{
  (void)in_sizes; (void)n_in; (void)out_size; (void)ws_size;
  const float* pt  = (const float*)d_in[0];
  const float* ang = (const float*)d_in[1];
  ConvW c0 = { (const float*)d_in[2],  (const float*)d_in[3],
               (const float*)d_in[4],  (const float*)d_in[5],
               (const float*)d_in[6],  (const float*)d_in[7] };
  ConvW c1 = { (const float*)d_in[8],  (const float*)d_in[9],
               (const float*)d_in[10], (const float*)d_in[11],
               (const float*)d_in[12], (const float*)d_in[13] };
  const float* R0  = (const float*)d_in[14];
  const float* rb0 = (const float*)d_in[15];
  const float* R1  = (const float*)d_in[16];
  const float* rb1 = (const float*)d_in[17];
  const float* R2  = (const float*)d_in[18];
  const float* rb2 = (const float*)d_in[19];
  float* gws = (float*)d_ws;   // 256 x 4 pooled features
  float* out = (float*)d_out;  // 256 x 3

  hipLaunchKernelGGL(fused_edgeconv_kernel, dim3(kB), dim3(256), 0, stream,
                     pt, ang, c0, c1, gws, out);
  hipLaunchKernelGGL(readout_kernel, dim3(1), dim3(512), 0, stream,
                     gws, R0, rb0, R1, rb1, R2, rb2, out);
}
